// GraphConstructor_5952824672352
// MI455X (gfx1250) — matrix operand order repaired and checked by reading, not compiled, not run on hardware
//
#include <hip/hip_runtime.h>
#include <hip/hip_bf16.h>
#include <math.h>

// Problem constants (from reference)
#define NN    8192
#define DIM   64
#define TOPK  32
#define ALPHAF 3.0f

typedef __attribute__((ext_vector_type(16))) _Float16 v16h;
typedef __attribute__((ext_vector_type(8)))  _Float16 v8h;
typedef __attribute__((ext_vector_type(8)))  float    v8f;

// ---------------------------------------------------------------------------
// Kernel 1: v1/v2 = tanh(ALPHA * (emb[idx] @ W^T + b)), stored f32 + f16.
// 8192*64 outputs per matrix; trivially compute bound, one thread per (r,d).
// ---------------------------------------------------------------------------
__global__ void compute_v_kernel(const int* __restrict__ idx,
                                 const float* __restrict__ emb1,
                                 const float* __restrict__ emb2,
                                 const float* __restrict__ w,
                                 const float* __restrict__ b,
                                 float* __restrict__ v1f, float* __restrict__ v2f,
                                 _Float16* __restrict__ v1h, _Float16* __restrict__ v2h) {
  int t = blockIdx.x * blockDim.x + threadIdx.x;   // [0, NN*DIM)
  int r = t >> 6;
  int d = t & 63;
  int src = idx[r];
  const float* e1 = emb1 + (size_t)src * DIM;
  const float* e2 = emb2 + (size_t)src * DIM;
  const float* wr = w + (size_t)d * DIM;           // lin1_w[d, :]  (nv @ W^T)
  float s1 = b[d], s2 = b[d];
#pragma unroll
  for (int k = 0; k < DIM; ++k) { s1 += e1[k] * wr[k]; s2 += e2[k] * wr[k]; }
  float t1 = tanhf(ALPHAF * s1);
  float t2 = tanhf(ALPHAF * s2);
  v1f[t] = t1;  v2f[t] = t2;
  v1h[t] = (_Float16)t1;  v2h[t] = (_Float16)t2;
}

// ---------------------------------------------------------------------------
// Kernel 2: fused  a = v1@v2^T - v2@v1^T  (WMMA f16->f32) + exact top-32.
// Block = 512 threads (16 waves) owns 16 rows. 16 panels x 512 cols.
// ---------------------------------------------------------------------------
__launch_bounds__(512, 1)
__global__ void fused_topk_kernel(const float* __restrict__ v1f,
                                  const float* __restrict__ v2f,
                                  const _Float16* __restrict__ v1h,
                                  const _Float16* __restrict__ v2h,
                                  float* __restrict__ outRows,
                                  float* __restrict__ outCols,
                                  float* __restrict__ outData) {
  __shared__ float panel[16][512];        // 32 KB: a-values for 16 rows x 512 cols
  __shared__ float selV[16][TOPK];        // extracted top-32 values per row
  __shared__ int   selC[16][TOPK];        // extracted top-32 columns per row

  const int wave  = threadIdx.x >> 5;     // 0..15
  const int lane  = threadIdx.x & 31;     // wave32
  const int m     = lane & 15;
  const int hi    = lane >> 4;            // lane half (WMMA operand striping)
  const int rbase = blockIdx.x * 16;

  // ---- Stage A operands in VGPRs: 16x32 f16 A-matrix layout (chunk c = K c*32..c*32+31)
  // lane(m,hi): halves 0..15 -> K = hi*16 + j
  v16h A1[2], A2[2];
#pragma unroll
  for (int c = 0; c < 2; ++c) {
    const _Float16* p1 = v1h + ((size_t)(rbase + m)) * DIM + c * 32;
    const _Float16* p2 = v2h + ((size_t)(rbase + m)) * DIM + c * 32;
    v8h lo1 = *(const v8h*)(p1 + hi * 16);
    v8h up1 = *(const v8h*)(p1 + hi * 16 + 8);
    v8h lo2 = *(const v8h*)(p2 + hi * 16);
    v8h up2 = *(const v8h*)(p2 + hi * 16 + 8);
    A1[c] = __builtin_shufflevector(lo1, up1, 0,1,2,3,4,5,6,7,8,9,10,11,12,13,14,15);
    A2[c] = __builtin_shufflevector(lo2, up2, 0,1,2,3,4,5,6,7,8,9,10,11,12,13,14,15);
  }

  // ---- Per-lane exact top-32 list (capacity 32 guarantees global top-32 coverage:
  // each lane sees a disjoint set of columns; any global top-32 element has <32
  // larger elements in its lane, so it survives in that lane's list).
  float lv[TOPK];
  int   lc[TOPK];
#pragma unroll
  for (int i = 0; i < TOPK; ++i) { lv[i] = -1e30f; lc[i] = 0; }
  float minv = -1e30f;
  int   minpos = 0;

  for (int p = 0; p < 16; ++p) {
    // Prefetch next panel's B rows (gfx1250 global_prefetch_b8 path).
    if (p + 1 < 16) {
      const _Float16* np2 = v2h + ((size_t)((p + 1) * 512 + threadIdx.x)) * DIM;
      const _Float16* np1 = v1h + ((size_t)((p + 1) * 512 + threadIdx.x)) * DIM;
      __builtin_prefetch(np2, 0, 0);
      __builtin_prefetch(np1, 0, 0);
    }

    // ---- Compute: wave handles 2 column tiles of this panel
#pragma unroll
    for (int t = 0; t < 2; ++t) {
      const int ltile = wave * 2 + t;              // 0..31 within panel
      const int jbase = p * 512 + ltile * 16;      // global column base
      v8f acc1 = {};                               // v1 @ v2^T tile
      v8f acc2 = {};                               // v2 @ v1^T tile
#pragma unroll
      for (int c = 0; c < 2; ++c) {
        // B operand 32x16 f16: lane(m,hi) holds column (jbase+m),
        // K = c*32 + hi*16 + (0..15)  -> 16 contiguous halves, 32B aligned.
        const _Float16* q2 = v2h + ((size_t)(jbase + m)) * DIM + c * 32 + hi * 16;
        const _Float16* q1 = v1h + ((size_t)(jbase + m)) * DIM + c * 32 + hi * 16;
        v16h B2 = *(const v16h*)q2;
        v16h B1 = *(const v16h*)q1;
        acc1 = __builtin_amdgcn_wmma_f32_16x16x32_f16(false, A1[c], false, B2,
                                                      (short)0, acc1, false, false);
        acc2 = __builtin_amdgcn_wmma_f32_16x16x32_f16(false, A2[c], false, B1,
                                                      (short)0, acc2, false, false);
      }
      // D layout: VGPR v, lanes0-15 -> (M=v, N=lane); lanes16-31 -> (M=v+8, N=lane-16)
      const int nloc = ltile * 16 + m;
#pragma unroll
      for (int v = 0; v < 8; ++v) {
        panel[v + hi * 8][nloc] = acc1[v] - acc2[v];   // antisymmetric score
      }
    }
    __syncthreads();

    // ---- Selection: wave w scans row w of the panel; lane covers cols lane (mod 32)
#pragma unroll
    for (int i = 0; i < 16; ++i) {
      const int cloc = lane + i * 32;
      const float val = panel[wave][cloc];
      const int col = p * 512 + cloc;
      if (val > minv) {
        // replace current min (fully unrolled selects keep arrays in VGPRs)
#pragma unroll
        for (int j = 0; j < TOPK; ++j) {
          const bool s = (j == minpos);
          lv[j] = s ? val : lv[j];
          lc[j] = s ? col : lc[j];
        }
        minv = lv[0]; minpos = 0;
#pragma unroll
        for (int j = 1; j < TOPK; ++j) {
          if (lv[j] < minv) { minv = lv[j]; minpos = j; }
        }
      }
    }
    __syncthreads();
  }

  // ---- Merge the 32 per-lane lists into the row's exact global top-32.
  for (int k = 0; k < TOPK; ++k) {
    float bm = -1e30f; int bp = 0;
#pragma unroll
    for (int j = 0; j < TOPK; ++j) { if (lv[j] > bm) { bm = lv[j]; bp = j; } }
    // wave32 argmax butterfly (all lanes converge on winner value + lane)
    float wv = bm; int wl = lane;
#pragma unroll
    for (int off = 16; off >= 1; off >>= 1) {
      float ov = __shfl_xor(wv, off, 32);
      int   ol = __shfl_xor(wl, off, 32);
      if (ov > wv || (ov == wv && ol < wl)) { wv = ov; wl = ol; }
    }
    if (lane == wl) {
      int bc = 0;
#pragma unroll
      for (int j = 0; j < TOPK; ++j) { if (j == bp) bc = lc[j]; }
      selV[wave][k] = bm;
      selC[wave][k] = bc;
#pragma unroll
      for (int j = 0; j < TOPK; ++j) { if (j == bp) lv[j] = -1e30f; }
    }
  }
  __syncthreads();

  // ---- Output: lane handles extraction slot `lane`; rank = sorted-by-column pos.
  const int myc = selC[wave][lane];
  int rank = 0;
#pragma unroll
  for (int j = 0; j < TOPK; ++j) rank += (selC[wave][j] < myc) ? 1 : 0;

  const int r = rbase + wave;
  // Exact f32 rescore of the selected entry (matches the f32 reference).
  const float* a1 = v1f + (size_t)r * DIM;
  const float* a2 = v2f + (size_t)r * DIM;
  const float* b1 = v1f + (size_t)myc * DIM;
  const float* b2 = v2f + (size_t)myc * DIM;
  float s1 = 0.f, s2 = 0.f;
#pragma unroll
  for (int kk = 0; kk < DIM; ++kk) { s1 += a1[kk] * b2[kk]; s2 += a2[kk] * b1[kk]; }
  float aval = s1 - s2;
  float adj = tanhf(ALPHAF * aval);
  adj = adj > 0.f ? adj : 0.f;

  const size_t base = (size_t)r * TOPK + rank;
  outRows[base] = (float)r;
  outCols[base] = (float)myc;
  outData[base] = adj;
}

// ---------------------------------------------------------------------------
// Launch. Inputs (setup_inputs order): idx, emb1, emb2, lin1_w, lin1_b,
// lin2_w, lin2_b (lin2 unused per reference). Output (float*): concat of
// index rows [N*K], index cols [N*K], data [N*K].
// Workspace: v1f, v2f (f32) + v1h, v2h (f16) = 6 MB.
// ---------------------------------------------------------------------------
extern "C" void kernel_launch(void* const* d_in, const int* in_sizes, int n_in,
                              void* d_out, int out_size, void* d_ws, size_t ws_size,
                              hipStream_t stream) {
  const int*   idx  = (const int*)  d_in[0];
  const float* emb1 = (const float*)d_in[1];
  const float* emb2 = (const float*)d_in[2];
  const float* w1   = (const float*)d_in[3];
  const float* b1   = (const float*)d_in[4];
  (void)in_sizes; (void)n_in; (void)out_size; (void)ws_size;

  char* ws = (char*)d_ws;
  float*    v1f = (float*)ws;                                       // 2 MB
  float*    v2f = (float*)(ws + (size_t)NN * DIM * 4);              // 2 MB
  _Float16* v1h = (_Float16*)(ws + (size_t)NN * DIM * 8);           // 1 MB
  _Float16* v2h = (_Float16*)(ws + (size_t)NN * DIM * 8 + (size_t)NN * DIM * 2); // 1 MB

  compute_v_kernel<<<(NN * DIM) / 256, 256, 0, stream>>>(
      idx, emb1, emb2, w1, b1, v1f, v2f, v1h, v2h);

  float* outRows = (float*)d_out;
  float* outCols = outRows + (size_t)NN * TOPK;
  float* outData = outRows + (size_t)2 * NN * TOPK;

  fused_topk_kernel<<<NN / 16, 512, 0, stream>>>(
      v1f, v2f, v1h, v2h, outRows, outCols, outData);
}
